// approx_Linear_int8_old_22900765622864
// MI455X (gfx1250) — 8-bit matrix operand order repaired and checked by reading, compiled, not run on hardware
//
#include <hip/hip_runtime.h>
#include <hip/hip_bf16.h>


#define N_DIM  512
#define K_DIM  512
#define O_DIM  512

typedef __attribute__((ext_vector_type(8))) int v8i;

#define WS_XQ_OFF  64
#define WS_WQ_OFF  (64 + N_DIM * K_DIM)

__global__ void init_ws_kernel(unsigned* umax) {
    if (threadIdx.x < 2) umax[threadIdx.x] = 0u;
}

__global__ void absmax_kernel(const float* __restrict__ p, int n,
                              unsigned* __restrict__ out) {
    float m = 0.0f;
    for (int i = blockIdx.x * blockDim.x + threadIdx.x; i < n;
         i += gridDim.x * blockDim.x) {
        m = fmaxf(m, fabsf(p[i]));
    }
    atomicMax(out, __float_as_uint(m));
}

__device__ __forceinline__ int quant1(float x, float inv) {
    float r = rintf(x * inv);
    r = fminf(fmaxf(r, -128.0f), 127.0f);
    return (int)r;
}

__global__ void quantize_kernel(const float* __restrict__ in, int n4,
                                const unsigned* __restrict__ umax, int which,
                                unsigned* __restrict__ outq) {
    const float mx  = __uint_as_float(umax[which]);
    const float inv = (mx > 0.0f) ? (127.0f / mx) : 0.0f;
    for (int i = blockIdx.x * blockDim.x + threadIdx.x; i < n4;
         i += gridDim.x * blockDim.x) {
        const float4 v = ((const float4*)in)[i];
        const int q0 = quant1(v.x, inv);
        const int q1 = quant1(v.y, inv);
        const int q2 = quant1(v.z, inv);
        const int q3 = quant1(v.w, inv);
        outq[i] = (unsigned)(q0 & 0xFF) | ((unsigned)(q1 & 0xFF) << 8) |
                  ((unsigned)(q2 & 0xFF) << 16) | ((unsigned)(q3 & 0xFF) << 24);
    }
}

__global__ __launch_bounds__(128) void gemm_iu8_wmma_kernel(
    const int* __restrict__ xq, const int* __restrict__ wq,
    const unsigned* __restrict__ umax, const float* __restrict__ bias,
    float* __restrict__ out) {
    const int lane = threadIdx.x & 31;
    const int wave = threadIdx.x >> 5;
    const int tile = blockIdx.x * 4 + wave;
    const int row_base = (tile >> 5) << 4;
    const int col_base = (tile & 31) << 4;

    const float sx = __uint_as_float(umax[0]) * (1.0f / 127.0f);
    const float sw = __uint_as_float(umax[1]) * (1.0f / 127.0f);
    const float scale = sx * sw;

    const int m = lane & 15;
    const int h = lane >> 4;

    v8i acc = {};
#pragma unroll
    for (int kb = 0; kb < 8; ++kb) {
        const int kbase = kb << 6;
        v8i a, b;
#pragma unroll
        for (int v = 0; v < 8; ++v) {
            const int ka = ((v >> 1) << 4) + ((v & 1) << 2) + (h << 3);
            a[v] = xq[(((row_base + m) * K_DIM) + kbase + ka) >> 2];
            const int kc = ((v >> 1) << 4) + ((v & 1) << 2) + (h << 3);
            b[v] = wq[(((col_base + m) * K_DIM) + kbase + kc) >> 2];
        }
        acc = __builtin_amdgcn_wmma_i32_16x16x64_iu8(
             true, a,  true, b, acc,
             false,  false);
    }

    const float bv = bias[col_base + m];
#pragma unroll
    for (int r = 0; r < 8; ++r) {
        const int row = row_base + r + (h << 3);
        out[row * O_DIM + col_base + m] = (float)acc[r] * scale + bv;
    }
}

extern "C" void kernel_launch(void* const* d_in, const int* in_sizes, int n_in,
                              void* d_out, int out_size, void* d_ws, size_t ws_size,
                              hipStream_t stream) {
    const float* x      = (const float*)d_in[0];
    const float* weight = (const float*)d_in[1];
    const float* bias   = (const float*)d_in[2];

    unsigned* umax = (unsigned*)d_ws;
    unsigned* xq   = (unsigned*)((char*)d_ws + WS_XQ_OFF);
    unsigned* wq   = (unsigned*)((char*)d_ws + WS_WQ_OFF);
    float*    out  = (float*)d_out;

    init_ws_kernel<<<1, 32, 0, stream>>>(umax);

    absmax_kernel<<<128, 256, 0, stream>>>(x,      N_DIM * K_DIM, &umax[0]);
    absmax_kernel<<<128, 256, 0, stream>>>(weight, O_DIM * K_DIM, &umax[1]);

    const int n4 = (N_DIM * K_DIM) / 4;
    quantize_kernel<<<n4 / 256, 256, 0, stream>>>(x,      n4, umax, 0, xq);
    quantize_kernel<<<n4 / 256, 256, 0, stream>>>(weight, n4, umax, 1, wq);

    gemm_iu8_wmma_kernel<<<256, 128, 0, stream>>>((const int*)xq, (const int*)wq,
                                                  umax, bias, out);
}
